// ODLUE_84782654423331
// MI455X (gfx1250) — matrix operand order repaired and checked by reading, not compiled, not run on hardware
//
#include <hip/hip_runtime.h>
#include <math.h>

// ---- problem constants (from reference) ----
#define N_DAYS  64
#define N_HOURS 24
#define NB      (N_DAYS * N_HOURS)   // 1536 rows
#define NL      500                  // links
#define NF      5                    // features
#define NP      20000                // paths
#define NOD     4000                 // OD pairs
#define KP1     512                  // link dim padded to multiple of 32 (WMMA K)
#define NL_PAD  512                  // output-link dim padded to multiple of 128
#define NP_PAD  20096                // path dim padded to multiple of 128 (157*128)

typedef __attribute__((ext_vector_type(16))) __bf16 v16bf;
typedef __attribute__((ext_vector_type(8)))  float  v8f;

static __device__ __forceinline__ __bf16 f2bf(float x) {
    // round-to-nearest-even f32 -> bf16
    unsigned u = __builtin_bit_cast(unsigned, x);
    unsigned r = u + 0x7fffu + ((u >> 16) & 1u);
    unsigned short h = (unsigned short)(r >> 16);
    return __builtin_bit_cast(__bf16, h);
}

// ---------------- prep kernels ----------------

// V[b][k] = sum_f X[b][k][f]*min(theta[f],0) + theta_links[k], bf16, K-padded with zeros
__global__ void prep_v_kernel(const float* __restrict__ X,
                              const float* __restrict__ theta,
                              const float* __restrict__ theta_links,
                              __bf16* __restrict__ Vbf) {
    int idx = blockIdx.x * blockDim.x + threadIdx.x;
    if (idx >= NB * KP1) return;
    int b = idx / KP1, k = idx % KP1;
    float v = 0.f;
    if (k < NL) {
        const float* xr = X + ((size_t)b * NL + k) * NF;
        #pragma unroll
        for (int f = 0; f < NF; ++f) v += xr[f] * fminf(theta[f], 0.f);
        v += theta_links[k];
    }
    Vbf[idx] = f2bf(k < NL ? v : 0.f);
}

// Dbf1[k][p] = bf16(D[k][p]) in (KP1 x NP_PAD), pad rows/cols zero
__global__ void conv_d1_kernel(const float* __restrict__ D, __bf16* __restrict__ Dbf1) {
    int idx = blockIdx.x * blockDim.x + threadIdx.x;
    if (idx >= KP1 * NP_PAD) return;
    int k = idx / NP_PAD, p = idx % NP_PAD;
    Dbf1[idx] = f2bf((k < NL && p < NP) ? D[(size_t)k * NP + p] : 0.f);
}

// Dbf2[p][l] = bf16(D[l][p]) (NP x NL_PAD, cols 500..511 zero), LDS-tiled transpose
__global__ void conv_d2_kernel(const float* __restrict__ D, __bf16* __restrict__ Dbf2) {
    __shared__ float tile[32][33];
    int pbase = blockIdx.x * 32;   // 625 tiles, exact
    int lbase = blockIdx.y * 32;   // 16 tiles, exact
    int tx = threadIdx.x, ty = threadIdx.y;   // block (32,8)
    #pragma unroll
    for (int j = 0; j < 4; ++j) {
        int l = lbase + ty + j * 8;
        tile[ty + j * 8][tx] = (l < NL) ? D[(size_t)l * NP + pbase + tx] : 0.f;
    }
    __syncthreads();
    #pragma unroll
    for (int j = 0; j < 4; ++j) {
        int p = pbase + ty + j * 8;
        Dbf2[(size_t)p * NL_PAD + lbase + tx] = f2bf(tile[tx][ty + j * 8]);
    }
}

// ---------------- segment offsets (seg_ids is sorted) ----------------

__global__ void seg_init_kernel(int* __restrict__ s, int* __restrict__ e) {
    int i = blockIdx.x * blockDim.x + threadIdx.x;
    if (i < NOD) { s[i] = 0; e[i] = 0; }   // empty groups -> empty range
}

__global__ void seg_mark_kernel(const int* __restrict__ seg,
                                int* __restrict__ s, int* __restrict__ e) {
    int p = blockIdx.x * blockDim.x + threadIdx.x;
    if (p >= NP) return;
    int g = seg[p];
    if (p == 0 || seg[p - 1] != g) s[g] = p;
    if (p == NP - 1 || seg[p + 1] != g) e[g] = p + 1;
}

// ---------------- fused segmented softmax + path flows ----------------
__global__ void softmax_flow_kernel(const float* __restrict__ VP,
                                    const int* __restrict__ s, const int* __restrict__ e,
                                    const float* __restrict__ q_sqrt,
                                    __bf16* __restrict__ Fbf) {
    int idx = blockIdx.x * blockDim.x + threadIdx.x;
    if (idx >= NB * NOD) return;
    int b = idx / NOD, od = idx % NOD;
    int p0 = s[od], p1 = e[od];
    if (p1 <= p0) return;
    const float* vr = VP + (size_t)b * NP_PAD;
    float m = -3.4e38f;
    for (int p = p0; p < p1; ++p) m = fmaxf(m, vr[p]);
    float sum = 0.f;
    for (int p = p0; p < p1; ++p) sum += __expf(vr[p] - m);
    float q = q_sqrt[od] * q_sqrt[od];
    float inv = q / sum;
    __bf16* fr = Fbf + (size_t)b * NP;
    for (int p = p0; p < p1; ++p) fr[p] = f2bf(__expf(vr[p] - m) * inv);
}

// ---------------- bf16 WMMA GEMM: C(f32) = A(bf16,MxK) * B(bf16,KxN) ----------------
// 256 threads = 8 waves; block tile 64(M) x 128(N); wave tile 32x32 (2x2 WMMA); K step 32.
// Requirements: M % 64 == 0, K % 32 == 0, N % 128 == 0 (callers pad buffers). Nout guards stores.
// LDS: As row-major [64][40]; Bt transposed [128 cols][40 ks]. Stride 40 => the 16 lanes of a
// ds_load_b128 fragment read hit distinct banks (20*r mod 64 injective), and fragments are
// contiguous: A lane chunks k=8h..8h+7 / 16+8h..23+8h ; B lane chunks the same as A.
__global__ __launch_bounds__(256)
void gemm_bf16f32(const __bf16* __restrict__ A, int lda,
                  const __bf16* __restrict__ B, int ldb,
                  float* __restrict__ C, int ldc,
                  int Nout, int ktiles) {
    __shared__ __align__(16) __bf16 As[64 * 40];
    __shared__ __align__(16) __bf16 Bt[128 * 40];

    const int t    = threadIdx.x;
    const int lane = t & 31;
    const int w    = t >> 5;        // wave 0..7
    const int wm   = w >> 2;        // 0..1  (M)
    const int wn   = w & 3;         // 0..3  (N)
    const int lrow = lane & 15;
    const int half = lane >> 4;

    const size_t mBlock = (size_t)blockIdx.y * 64;
    const size_t nBlock = (size_t)blockIdx.x * 128;

    // A staging: thread -> row t>>2 (0..63), col (t&3)*8 : one 16B chunk
    const int arow = t >> 2;
    const int acol = (t & 3) * 8;
    // B staging: thread -> k-pair t>>4 (0..15) i.e. rows 2kp,2kp+1 ; col chunk (t&15)*8
    const int kp   = t >> 4;
    const int bcol = (t & 15) * 8;

    const __bf16* aSrc  = A + (mBlock + arow) * lda + acol;
    const __bf16* bSrc0 = B + (size_t)(2 * kp) * ldb + nBlock + bcol;
    const __bf16* bSrc1 = bSrc0 + ldb;

    v8f acc[2][2] = {};

    struct Q2 { uint4 a, b; };

    for (int kt = 0; kt < ktiles; ++kt) {
        // stage A tile 64x32 (row-major, stride 40)
        *(uint4*)(&As[arow * 40 + acol]) = *(const uint4*)aSrc;
        // stage B tile 32x128 transposed: pack (k, k+1) per column -> one b32 store
        uint4 r0 = *(const uint4*)bSrc0;
        uint4 r1 = *(const uint4*)bSrc1;
        const unsigned short* s0 = (const unsigned short*)&r0;
        const unsigned short* s1 = (const unsigned short*)&r1;
        #pragma unroll
        for (int i = 0; i < 8; ++i) {
            unsigned pk = (unsigned)s0[i] | ((unsigned)s1[i] << 16);
            *(unsigned*)(&Bt[(bcol + i) * 40 + 2 * kp]) = pk;
        }
        aSrc  += 32;
        bSrc0 += (size_t)32 * ldb;
        bSrc1 += (size_t)32 * ldb;
        if (kt + 1 < ktiles) {
            __builtin_prefetch(aSrc, 0, 3);
            __builtin_prefetch(bSrc0, 0, 3);
        }
        __syncthreads();

        v16bf afrag[2], bfrag[2];
        #pragma unroll
        for (int tm = 0; tm < 2; ++tm) {
            const __bf16* ab = &As[(32 * wm + 16 * tm + lrow) * 40 + 8 * half];
            Q2 q;
            q.a = *(const uint4*)(ab);        // elements 0..7  : k = 8h + 0..7
            q.b = *(const uint4*)(ab + 16);   // elements 8..15 : k = 16 + 8h + 0..7
            afrag[tm] = __builtin_bit_cast(v16bf, q);
        }
        #pragma unroll
        for (int tn = 0; tn < 2; ++tn) {
            const __bf16* bb = &Bt[(32 * wn + 16 * tn + lrow) * 40 + 8 * half];
            Q2 q;
            q.a = *(const uint4*)(bb);        // elements 0..7  : k = 8h + 0..7
            q.b = *(const uint4*)(bb + 16);   // elements 8..15 : k = 16 + 8h + 0..7
            bfrag[tn] = __builtin_bit_cast(v16bf, q);
        }
        #pragma unroll
        for (int tm = 0; tm < 2; ++tm)
            #pragma unroll
            for (int tn = 0; tn < 2; ++tn)
                acc[tm][tn] = __builtin_amdgcn_wmma_f32_16x16x32_bf16(
                    false, afrag[tm], false, bfrag[tn],
                    (short)0, acc[tm][tn], false, false);
        __syncthreads();
    }

    // store C fragments (16x16 f32 layout: VGPR r -> M = r + 8*half, N = lane%16)
    #pragma unroll
    for (int tm = 0; tm < 2; ++tm) {
        #pragma unroll
        for (int tn = 0; tn < 2; ++tn) {
            int col = (int)nBlock + 32 * wn + 16 * tn + lrow;
            if (col < Nout) {
                #pragma unroll
                for (int r = 0; r < 8; ++r) {
                    size_t row = mBlock + 32 * wm + 16 * tm + r + 8 * half;
                    C[row * ldc + col] = acc[tm][tn][r];
                }
            }
        }
    }
}

// ---------------- host-side orchestration ----------------

extern "C" void kernel_launch(void* const* d_in, const int* in_sizes, int n_in,
                              void* d_out, int out_size, void* d_ws, size_t ws_size,
                              hipStream_t stream) {
    (void)in_sizes; (void)n_in; (void)out_size; (void)ws_size;

    const float* X           = (const float*)d_in[0];
    const float* theta       = (const float*)d_in[1];
    const float* theta_links = (const float*)d_in[2];
    const float* q_sqrt      = (const float*)d_in[3];
    const float* D           = (const float*)d_in[4];
    const int*   seg         = (const int*)d_in[5];
    float*       out         = (float*)d_out;

    char* ws = (char*)d_ws;
    size_t off = 0;
    auto alloc = [&](size_t bytes) -> char* {
        char* p = ws + off;
        off += (bytes + 255) & ~(size_t)255;
        return p;
    };
    __bf16* Vbf  = (__bf16*)alloc((size_t)NB * KP1 * 2);        //   1.6 MB
    __bf16* Dbf1 = (__bf16*)alloc((size_t)KP1 * NP_PAD * 2);    //  20.6 MB
    __bf16* Dbf2 = (__bf16*)alloc((size_t)NP * NL_PAD * 2);     //  20.5 MB
    float*  VP   = (float*) alloc((size_t)NB * NP_PAD * 4);     // 123.5 MB
    __bf16* Fbf  = (__bf16*)alloc((size_t)NB * NP * 2);         //  61.4 MB
    int*    segS = (int*)   alloc((size_t)NOD * 4);
    int*    segE = (int*)   alloc((size_t)NOD * 4);

    prep_v_kernel<<<(NB * KP1 + 255) / 256, 256, 0, stream>>>(X, theta, theta_links, Vbf);
    conv_d1_kernel<<<(KP1 * NP_PAD + 255) / 256, 256, 0, stream>>>(D, Dbf1);
    conv_d2_kernel<<<dim3(NP / 32, NL_PAD / 32), dim3(32, 8), 0, stream>>>(D, Dbf2);
    seg_init_kernel<<<(NOD + 255) / 256, 256, 0, stream>>>(segS, segE);
    seg_mark_kernel<<<(NP + 255) / 256, 256, 0, stream>>>(seg, segS, segE);

    // GEMM1: VP(1536 x 20096) = Vbf(1536 x 512) * Dbf1(512 x 20096)
    gemm_bf16f32<<<dim3(NP_PAD / 128, NB / 64), 256, 0, stream>>>(
        Vbf, KP1, Dbf1, NP_PAD, VP, NP_PAD, /*Nout=*/NP_PAD, /*ktiles=*/KP1 / 32);

    softmax_flow_kernel<<<(NB * NOD + 255) / 256, 256, 0, stream>>>(VP, segS, segE, q_sqrt, Fbf);

    // GEMM2: out(1536 x 500) = Fbf(1536 x 20000) * Dbf2(20000 x 512), stores guarded to 500
    gemm_bf16f32<<<dim3(NL_PAD / 128, NB / 64), 256, 0, stream>>>(
        Fbf, NP, Dbf2, NL_PAD, out, NL, /*Nout=*/NL, /*ktiles=*/NP / 32);
}
